// MaskSerialThorMoE_7533372638011
// MI455X (gfx1250) — matrix operand order repaired and checked by reading, not compiled, not run on hardware
//
#include <hip/hip_runtime.h>
#include <math.h>

#define T_TOK 8192
#define H_DIM 768
#define I_DIM 3072
#define E_EXP 8

typedef _Float16 h16;
typedef __attribute__((ext_vector_type(16))) _Float16 v16h;
typedef __attribute__((ext_vector_type(8)))  _Float16 v8h;
typedef __attribute__((ext_vector_type(8)))  float    v8f;
typedef int v4i __attribute__((vector_size(16)));   // matches async builtin param type

#if __has_builtin(__builtin_amdgcn_global_load_async_to_lds_b128)
#define HAVE_ASYNC_LDS 1
#else
#define HAVE_ASYNC_LDS 0
#endif

// stage one 16B chunk: global -> LDS (async on CDNA5, sync fallback otherwise)
__device__ __forceinline__ void stage16(const h16* g, h16* l) {
#if HAVE_ASYNC_LDS
  __builtin_amdgcn_global_load_async_to_lds_b128(
      (__attribute__((address_space(1))) v4i*)(h16*)g,
      (__attribute__((address_space(3))) v4i*)l, 0, 0);
#else
  *(v8h*)l = *(const v8h*)g;
#endif
}

__device__ __forceinline__ void wait_async_le4() {
#if HAVE_ASYNC_LDS
#if __has_builtin(__builtin_amdgcn_s_wait_asynccnt)
  __builtin_amdgcn_s_wait_asynccnt(4);
#else
  asm volatile("s_wait_asynccnt 0x4" ::: "memory");
#endif
#endif
}

__device__ __forceinline__ void wait_async_le0() {
#if HAVE_ASYNC_LDS
#if __has_builtin(__builtin_amdgcn_s_wait_asynccnt)
  __builtin_amdgcn_s_wait_asynccnt(0);
#else
  asm volatile("s_wait_asynccnt 0x0" ::: "memory");
#endif
#endif
}

__device__ __forceinline__ v16h join8(v8h lo, v8h hi) {
  return __builtin_shufflevector(lo, hi, 0,1,2,3,4,5,6,7,8,9,10,11,12,13,14,15);
}

// ---------------- routing ----------------
__global__ void route_zero(int* counts) {
  if (threadIdx.x < E_EXP) counts[threadIdx.x] = 0;
}

__global__ void route_count(const int* __restrict__ mask, int* __restrict__ counts) {
  int t = blockIdx.x * 256 + threadIdx.x;
  if (t < T_TOK) atomicAdd(&counts[mask[t]], 1);
}

__global__ void route_scan(const int* __restrict__ counts, int* __restrict__ offs,
                           int* __restrict__ cursor) {
  if (threadIdx.x == 0) {
    int a = 0;
    for (int e = 0; e < E_EXP; ++e) { offs[e] = a; cursor[e] = a; a += counts[e]; }
  }
}

__global__ void route_scatter(const int* __restrict__ mask, int* __restrict__ cursor,
                              int* __restrict__ perm) {
  int t = blockIdx.x * 256 + threadIdx.x;
  if (t < T_TOK) {
    int e = mask[t];
    int pos = atomicAdd(&cursor[e], 1);
    perm[pos] = t;
  }
}

// ---------------- precision conversion ----------------
__global__ void cast_hidden(const float* __restrict__ in, h16* __restrict__ out, int n) {
  int i = blockIdx.x * 256 + threadIdx.x;
  if (i < n) out[i] = (h16)in[i];
}

// in: (E, R, C) fp32  ->  out: (E, C, R) fp16   (transpose within each expert)
__global__ __launch_bounds__(256)
void transpose_cast(const float* __restrict__ in, h16* __restrict__ out, int R, int C) {
  __shared__ float tile[32][33];
  const int e  = blockIdx.z;
  const int c0 = blockIdx.x * 32;
  const int r0 = blockIdx.y * 32;
  const int tx = threadIdx.x;          // 0..31
  const int ty = threadIdx.y;          // 0..7
  const float* src = in  + (size_t)e * R * C;
  h16*         dst = out + (size_t)e * R * C;
  #pragma unroll
  for (int k = 0; k < 32; k += 8)
    tile[ty + k][tx] = src[(size_t)(r0 + ty + k) * C + (c0 + tx)];
  __syncthreads();
  #pragma unroll
  for (int k = 0; k < 32; k += 8)
    dst[(size_t)(c0 + ty + k) * R + (r0 + tx)] = (h16)tile[tx][ty + k];
}

// ---------------- fused expert GEMM ----------------
// Block tile: M=64 gathered tokens x N=128 columns, 4 waves.
// Each wave: two 16-wide N-subtiles x four 16-row M-subtiles = 8 WMMA accumulators.
// K is consumed in 64-wide chunks, double-buffered in LDS via async-to-LDS loads.
// MODE 1 : out = fp16 gelu(acc + bias)          -> outh (row stride NTOT)
// MODE 2 : out = fp32 acc + bias + hidden resid -> outf (row stride NTOT)
template<int KTOT, int NTOT, int MODE>
__global__ __launch_bounds__(128)
void moe_gemm(const h16* __restrict__ A, const h16* __restrict__ Bt,
              const float* __restrict__ bias,
              const int* __restrict__ perm, const int* __restrict__ offs,
              const int* __restrict__ cnts,
              const float* __restrict__ hidden,
              h16* __restrict__ outh, float* __restrict__ outf) {
  const int e   = blockIdx.z;
  const int mt  = blockIdx.y;
  const int nt  = blockIdx.x;
  const int cnt = cnts[e];
  const int row0 = mt * 64;
  if (row0 >= cnt) return;                      // uniform per block: EXEC stays full
  const int base = offs[e];
  int rows_valid = cnt - row0; if (rows_valid > 64) rows_valid = 64;

  __shared__ h16 As[2][64][72];                 // ping-pong 64x64 K-chunks (+8 pad)
  __shared__ int toks[64];

  const int tid = threadIdx.x;
  if (tid < 64) {
    int r = tid < rows_valid ? tid : rows_valid - 1;   // clamp padding rows
    toks[tid] = perm[base + row0 + r];
  }
  __syncthreads();

  const int lane  = tid & 31;
  const int wave  = tid >> 5;
  const int nloc  = lane & 15;
  const int khalf = lane >> 4;                  // 0 or 1
  const int n0    = nt * 128 + wave * 32 + nloc;

  // B fragment bases: lane column n, K-half khalf (16 contiguous fp16 = one v16h)
  const h16* Bb0 = Bt + ((size_t)e * NTOT + n0) * KTOT + khalf * 16;
  const h16* Bb1 = Bb0 + (size_t)16 * KTOT;     // second N-subtile (n0+16)

  // hoisted per-thread staging slots (4 x 16B per thread per chunk)
  int rr[4], cc[4];
  const h16* ga[4];
  #pragma unroll
  for (int i = 0; i < 4; ++i) {
    int idx = tid + i * 128;                    // 512 chunks = 64 rows x 8 col-chunks
    rr[i] = idx >> 3;
    cc[i] = (idx & 7) * 8;
    ga[i] = A + (size_t)toks[rr[i]] * KTOT + cc[i];
  }

  // prologue: stage chunk 0 into buffer 0
  #pragma unroll
  for (int i = 0; i < 4; ++i) stage16(ga[i], &As[0][rr[i]][cc[i]]);

  v8f acc[4][2] = {};
  constexpr int KC = KTOT / 64;

  for (int kc = 0; kc < KC; ++kc) {
    const int cur = kc & 1;
    if (kc + 1 < KC) {
      __syncthreads();                          // all waves done reading buf[cur^1]
      #pragma unroll
      for (int i = 0; i < 4; ++i)
        stage16(ga[i] + (kc + 1) * 64, &As[cur ^ 1][rr[i]][cc[i]]);
      wait_async_le4();                         // chunk kc's batch retired (in-order)
    } else {
      wait_async_le0();
    }
    __syncthreads();                            // chunk kc visible to all waves

    #pragma unroll
    for (int ks = 0; ks < 64; ks += 32) {
      v16h b0 = *(const v16h*)(Bb0 + kc * 64 + ks);
      v16h b1 = *(const v16h*)(Bb1 + kc * 64 + ks);
      #pragma unroll
      for (int ms = 0; ms < 4; ++ms) {
        const h16* ap = &As[cur][ms * 16 + nloc][ks + khalf * 16];
        v8h alo = *(const v8h*)ap;              // K = kb..kb+7
        v8h ahi = *(const v8h*)(ap + 8);        // K = kb+8..kb+15
        v16h afrag = join8(alo, ahi);
        acc[ms][0] = __builtin_amdgcn_wmma_f32_16x16x32_f16(
            false, afrag, false, b0, (short)0, acc[ms][0], false, false);
        acc[ms][1] = __builtin_amdgcn_wmma_f32_16x16x32_f16(
            false, afrag, false, b1, (short)0, acc[ms][1], false, false);
      }
    }
  }

  // epilogue: D layout -> lane: n = lane&15; vgpr j: m = j + 8*(lane>>4)
  #pragma unroll
  for (int ms = 0; ms < 4; ++ms) {
    #pragma unroll
    for (int nn = 0; nn < 2; ++nn) {
      int n = n0 + nn * 16;
      #pragma unroll
      for (int j = 0; j < 8; ++j) {
        int m = ms * 16 + j + khalf * 8;
        if (m < rows_valid) {
          int tok = toks[m];
          float v = acc[ms][nn][j] + bias[e * NTOT + n];
          if (MODE == 1) {
            v = 0.5f * v * (1.0f + erff(v * 0.70710678118654752f));
            outh[(size_t)tok * NTOT + n] = (h16)v;
          } else {
            v += hidden[(size_t)tok * NTOT + n];
            outf[(size_t)tok * NTOT + n] = v;
          }
        }
      }
    }
  }
}

// ---------------- in-place LayerNorm (one wave32 per token row) ----------------
__global__ __launch_bounds__(256)
void ln_kernel(float* __restrict__ x, const float* __restrict__ gamma,
               const float* __restrict__ beta) {
  const int wave = threadIdx.x >> 5;
  const int lane = threadIdx.x & 31;
  const int row  = blockIdx.x * 8 + wave;
  if (row >= T_TOK) return;
  float* p = x + (size_t)row * H_DIM;
  float vals[H_DIM / 32];
  float s = 0.f, s2 = 0.f;
  #pragma unroll
  for (int i = 0; i < H_DIM / 32; ++i) {
    float v = p[lane + i * 32];
    vals[i] = v; s += v; s2 += v * v;
  }
  #pragma unroll
  for (int o = 16; o > 0; o >>= 1) {
    s  += __shfl_xor(s,  o, 32);
    s2 += __shfl_xor(s2, o, 32);
  }
  const float mu  = s * (1.0f / H_DIM);
  const float var = s2 * (1.0f / H_DIM) - mu * mu;
  const float inv = rsqrtf(var + 1e-12f);
  #pragma unroll
  for (int i = 0; i < H_DIM / 32; ++i) {
    int c = lane + i * 32;
    p[c] = (vals[i] - mu) * inv * gamma[c] + beta[c];
  }
}

// ---------------- launcher ----------------
extern "C" void kernel_launch(void* const* d_in, const int* in_sizes, int n_in,
                              void* d_out, int out_size, void* d_ws, size_t ws_size,
                              hipStream_t stream) {
  const float* hs    = (const float*)d_in[0];   // (1,T,H)
  const int*   mask  = (const int*)d_in[1];     // (1,T)
  const float* w1    = (const float*)d_in[2];   // (E,H,I)
  const float* b1    = (const float*)d_in[3];   // (E,I)
  const float* w2    = (const float*)d_in[4];   // (E,I,H)
  const float* b2    = (const float*)d_in[5];   // (E,H)
  const float* gam   = (const float*)d_in[6];   // (H)
  const float* bet   = (const float*)d_in[7];   // (H)
  float* out = (float*)d_out;                   // (1,T,H) fp32

  // workspace carve-up (~138 MB)
  char* w = (char*)d_ws;
  h16* hid_h  = (h16*)w;  w += (size_t)T_TOK * H_DIM * sizeof(h16);
  h16* w1t    = (h16*)w;  w += (size_t)E_EXP * I_DIM * H_DIM * sizeof(h16);
  h16* w2t    = (h16*)w;  w += (size_t)E_EXP * H_DIM * I_DIM * sizeof(h16);
  h16* interh = (h16*)w;  w += (size_t)T_TOK * I_DIM * sizeof(h16);
  int* counts = (int*)w;  w += 64;
  int* offs   = (int*)w;  w += 64;
  int* cursor = (int*)w;  w += 64;
  int* perm   = (int*)w;  w += (size_t)T_TOK * sizeof(int);

  // 1) routing
  route_zero<<<1, 32, 0, stream>>>(counts);
  route_count<<<T_TOK / 256, 256, 0, stream>>>(mask, counts);
  route_scan<<<1, 32, 0, stream>>>(counts, offs, cursor);
  route_scatter<<<T_TOK / 256, 256, 0, stream>>>(mask, cursor, perm);

  // 2) fp32 -> fp16 (weights transposed to N-major for WMMA B fragments)
  cast_hidden<<<(T_TOK * H_DIM) / 256, 256, 0, stream>>>(hs, hid_h, T_TOK * H_DIM);
  transpose_cast<<<dim3(I_DIM / 32, H_DIM / 32, E_EXP), dim3(32, 8), 0, stream>>>(
      w1, w1t, H_DIM, I_DIM);   // (E,H,I) -> (E,I,H)
  transpose_cast<<<dim3(H_DIM / 32, I_DIM / 32, E_EXP), dim3(32, 8), 0, stream>>>(
      w2, w2t, I_DIM, H_DIM);   // (E,I,H) -> (E,H,I)

  // 3) GEMM1 + GELU  (K=H, N=I)
  moe_gemm<H_DIM, I_DIM, 1><<<dim3(I_DIM / 128, T_TOK / 64, E_EXP), 128, 0, stream>>>(
      hid_h, w1t, b1, perm, offs, counts, nullptr, interh, nullptr);

  // 4) GEMM2 + bias + residual -> d_out (fp32)  (K=I, N=H)
  moe_gemm<I_DIM, H_DIM, 2><<<dim3(H_DIM / 128, T_TOK / 64, E_EXP), 128, 0, stream>>>(
      interh, w2t, b2, perm, offs, counts, hs, nullptr, out);

  // 5) in-place LayerNorm
  ln_kernel<<<T_TOK / 8, 256, 0, stream>>>(out, gam, bet);
}
